// Attention2_4801773437148
// MI455X (gfx1250) — compile-verified, matrix operand order checked by reading, not run on hardware
//
#include <hip/hip_runtime.h>
#include <hip/hip_bf16.h>
#include <cstdint>

typedef __attribute__((ext_vector_type(16))) _Float16 v16h;
typedef __attribute__((ext_vector_type(8)))  _Float16 v8h;
typedef __attribute__((ext_vector_type(8)))  float    v8f;
typedef __attribute__((ext_vector_type(4)))  float    v4f;
typedef __attribute__((ext_vector_type(2)))  float    v2f;

#define HEADS 8
#define DH    64
#define NTOK  4096
#define BATCH 4
#define BHD   32          // BATCH*HEADS
#define LMK   256         // landmarks
#define DIMF  512
#define KERN  33

// ---------------------------------------------------------------------------
// CDNA5 async global->LDS copy (ASYNCcnt-tracked, bypasses VGPRs).
// Per-lane: LDS[vdst[lane]] = MEM[vaddr[lane]], 16 bytes.  Generic pointers to
// LDS carry the LDS byte offset in their low 32 bits (aperture mapping).
// ---------------------------------------------------------------------------
__device__ __forceinline__ void async_ld_b128(void* lds_ptr, const void* gptr) {
  uint32_t l = (uint32_t)(uintptr_t)lds_ptr;
  asm volatile("global_load_async_to_lds_b128 %0, %1, off"
               :: "v"(l), "v"(gptr) : "memory");
}
__device__ __forceinline__ void wait_async0() {
  asm volatile("s_wait_asynccnt 0x0" ::: "memory");
}

// f16 WMMA fragment from LDS tile with row stride 32 halves.
// ISA 16-bit A layout: m = lane&15, kbase = lane<16?0:8; K kbase+0..7 then +16.
// B tiles are stored [N][K] so the same loader applies (n = lane&15).
__device__ __forceinline__ v16h ldfrag16(const _Float16* tile, int row0, int lane) {
  int m  = lane & 15;
  int kb = (lane >> 4) << 3;
  const _Float16* p = tile + (row0 + m) * 32 + kb;
  v8h lo = *(const v8h*)p;
  v8h hi = *(const v8h*)(p + 16);
  v16h f;
#pragma unroll
  for (int i = 0; i < 8; ++i) { f[i] = lo[i]; f[i + 8] = hi[i]; }
  return f;
}

// ===========================================================================
// Batched f16 GEMM:  C[z] = A[z] (MxK, row-major) * B[z]^T  (B stored NxK).
// 256 threads = 8 waves (4 M x 2 N), wave tile 32x32, block tile 128x64,
// K step 32, double-buffered LDS staged entirely with async b128 copies.
// MODE: 0 f16 store | 1 f32 store | 2 f32 + bias[n] + resid (batch 1)
//       3 QKV split (q*scale / k / v + v-transposed) | 4 f16 transposed store
// ===========================================================================
template <int MODE>
__global__ __launch_bounds__(256) void gemm_f16_k(
    const _Float16* __restrict__ A, const _Float16* __restrict__ B,
    int M, int N, int K,
    long sAb, long sBb, long sOb,
    void* __restrict__ Out, const float* __restrict__ bias,
    const float* __restrict__ resid,
    _Float16* __restrict__ q_out, _Float16* __restrict__ k_out,
    _Float16* __restrict__ v_out, _Float16* __restrict__ vt_out, float qscale)
{
  __shared__ __align__(16) _Float16 sA[2][128 * 32];
  __shared__ __align__(16) _Float16 sB[2][64 * 32];
  int z = blockIdx.z;
  const _Float16* Ab = A + (long)z * sAb;
  const _Float16* Bb = B + (long)z * sBb;
  int t = threadIdx.x, lane = t & 31, wv = t >> 5;
  int wm = wv >> 1, wn = wv & 1;
  int bm0 = blockIdx.y * 128, bn0 = blockIdx.x * 64;

  int arow = t >> 1, aseg = (t & 1) << 4;     // A: 2 async chunks / thread
  int bnn  = t >> 2, bks  = (t & 3) << 3;     // B: 1 async chunk / thread
  auto load_tiles = [&](int k0, int buf) {
    const _Float16* ga = Ab + (long)(bm0 + arow) * K + k0 + aseg;
    async_ld_b128(&sA[buf][arow * 32 + aseg],     ga);
    async_ld_b128(&sA[buf][arow * 32 + aseg + 8], ga + 8);
    async_ld_b128(&sB[buf][bnn * 32 + bks],
                  Bb + (long)(bn0 + bnn) * K + k0 + bks);
  };

  v8f acc[2][2] = {};
  int buf = 0;
  load_tiles(0, 0);
  for (int k0 = 0; k0 < K; k0 += 32) {
    wait_async0();
    __syncthreads();
    if (k0 + 32 < K) load_tiles(k0 + 32, buf ^ 1);   // prefetch overlaps WMMA
    v16h a0 = ldfrag16(sA[buf], wm * 32,      lane);
    v16h a1 = ldfrag16(sA[buf], wm * 32 + 16, lane);
    v16h b0 = ldfrag16(sB[buf], wn * 32,      lane);
    v16h b1 = ldfrag16(sB[buf], wn * 32 + 16, lane);
    acc[0][0] = __builtin_amdgcn_wmma_f32_16x16x32_f16(false, a0, false, b0, (short)0, acc[0][0], false, false);
    acc[0][1] = __builtin_amdgcn_wmma_f32_16x16x32_f16(false, a0, false, b1, (short)0, acc[0][1], false, false);
    acc[1][0] = __builtin_amdgcn_wmma_f32_16x16x32_f16(false, a1, false, b0, (short)0, acc[1][0], false, false);
    acc[1][1] = __builtin_amdgcn_wmma_f32_16x16x32_f16(false, a1, false, b1, (short)0, acc[1][1], false, false);
    buf ^= 1;
  }
  // C layout: VGPR r -> M = r + (lane>=16?8:0), N = lane&15
  int nl = lane & 15, moff = (lane >> 4) << 3;
#pragma unroll
  for (int i = 0; i < 2; ++i)
#pragma unroll
    for (int j = 0; j < 2; ++j) {
      int gm0 = bm0 + wm * 32 + i * 16;
      int gn0 = bn0 + wn * 32 + j * 16;
#pragma unroll
      for (int r = 0; r < 8; ++r) {
        int gm = gm0 + moff + r, gn = gn0 + nl;
        float val = acc[i][j][r];
        long oidx = (long)gm * N + gn;
        if (MODE == 0) {
          ((_Float16*)Out)[(long)z * sOb + oidx] = (_Float16)val;
        } else if (MODE == 1) {
          ((float*)Out)[(long)z * sOb + oidx] = val;
        } else if (MODE == 2) {
          ((float*)Out)[oidx] = val + bias[gn] + resid[oidx];
        } else if (MODE == 3) {             // QKV split, M=b*n, N=1536
          int sel = gn >> 9, hh = (gn >> 6) & 7, dd = gn & 63;
          int bb = gm >> 12, tok = gm & 4095;
          long zi = (long)(bb * HEADS + hh);
          long di = (zi * NTOK + tok) * DH + dd;
          if (sel == 0)      q_out[di] = (_Float16)(val * qscale);
          else if (sel == 1) k_out[di] = (_Float16)val;
          else {
            v_out[di] = (_Float16)val;
            vt_out[(zi * DH + dd) * NTOK + tok] = (_Float16)val;  // V^T for attn3@v
          }
        } else {                            // 4: transposed f16 store
          ((_Float16*)Out)[(long)z * sOb + (long)gn * M + gm] = (_Float16)val;
        }
      }
    }
}

// ===========================================================================
// Batched f32 GEMM for the Newton-Schulz pinv chain (256x256 matrices).
// B supplied TRANSPOSED (NxK) so both tiles stage via async b128 copies.
// block tile 64x128, 8 waves (2 M x 4 N), v_wmma_f32_16x16x4_f32,
// double-buffered LDS.  Out = beta*acc + gamma*Cin ; Out and/or OutT optional.
// ===========================================================================
__global__ __launch_bounds__(256) void gemm_f32_k(
    const float* __restrict__ A, const float* __restrict__ BT,
    const float* __restrict__ Cin, float* __restrict__ Out,
    float* __restrict__ OutT, float beta, float gamma)
{
  __shared__ __align__(16) float sA[2][64 * 16];
  __shared__ __align__(16) float sB[2][128 * 16];    // [n][k]
  long zo = (long)blockIdx.z * 65536;
  const float* Ab = A + zo;
  const float* Bb = BT + zo;
  const float* Cb = Cin ? Cin + zo : nullptr;
  int t = threadIdx.x, lane = t & 31, wv = t >> 5;
  int wm = wv >> 2, wn = wv & 3;
  int bm0 = blockIdx.y * 64, bn0 = blockIdx.x * 128;

  int arow = t >> 2, aseg = (t & 3) << 2;            // A: 1 chunk / thread
  int c0 = t << 1;                                   // B: 2 chunks / thread
  auto load_tiles = [&](int k0, int buf) {
    async_ld_b128(&sA[buf][arow * 16 + aseg],
                  Ab + (long)(bm0 + arow) * 256 + k0 + aseg);
#pragma unroll
    for (int c = 0; c < 2; ++c) {
      int nn = (c0 + c) >> 2, ks = ((c0 + c) & 3) << 2;
      async_ld_b128(&sB[buf][nn * 16 + ks],
                    Bb + (long)(bn0 + nn) * 256 + k0 + ks);
    }
  };

  v8f acc[2][2] = {};
  int buf = 0;
  load_tiles(0, 0);
  for (int k0 = 0; k0 < 256; k0 += 16) {
    wait_async0();
    __syncthreads();
    if (k0 + 16 < 256) load_tiles(k0 + 16, buf ^ 1);
    int m = lane & 15, kb2 = (lane >> 4) << 1;       // f32 16x4 frag: 2 floats
#pragma unroll
    for (int kb = 0; kb < 16; kb += 4) {
      v2f a0 = *(const v2f*)&sA[buf][(wm * 32 + m) * 16 + kb + kb2];
      v2f a1 = *(const v2f*)&sA[buf][(wm * 32 + 16 + m) * 16 + kb + kb2];
      v2f b0 = *(const v2f*)&sB[buf][(wn * 32 + m) * 16 + kb + kb2];
      v2f b1 = *(const v2f*)&sB[buf][(wn * 32 + 16 + m) * 16 + kb + kb2];
      acc[0][0] = __builtin_amdgcn_wmma_f32_16x16x4_f32(false, a0, false, b0, (short)0, acc[0][0], false, false);
      acc[0][1] = __builtin_amdgcn_wmma_f32_16x16x4_f32(false, a0, false, b1, (short)0, acc[0][1], false, false);
      acc[1][0] = __builtin_amdgcn_wmma_f32_16x16x4_f32(false, a1, false, b0, (short)0, acc[1][0], false, false);
      acc[1][1] = __builtin_amdgcn_wmma_f32_16x16x4_f32(false, a1, false, b1, (short)0, acc[1][1], false, false);
    }
    buf ^= 1;
  }
  int nl = lane & 15, moff = (lane >> 4) << 3;
#pragma unroll
  for (int i = 0; i < 2; ++i)
#pragma unroll
    for (int j = 0; j < 2; ++j)
#pragma unroll
      for (int r = 0; r < 8; ++r) {
        int gm = bm0 + wm * 32 + i * 16 + moff + r;
        int gn = bn0 + wn * 32 + j * 16 + nl;
        long oi = (long)gm * 256 + gn;
        float v = beta * acc[i][j][r];
        if (Cb) v += gamma * Cb[oi];
        if (Out)  (Out + zo)[oi] = v;
        if (OutT) (OutT + zo)[(long)gn * 256 + gm] = v;
      }
}

// ======================= small support kernels =============================
__global__ __launch_bounds__(256) void layernorm_k(
    const float* __restrict__ x, const float* __restrict__ g,
    const float* __restrict__ b, _Float16* __restrict__ out)
{
  __shared__ float red[256];
  long row = blockIdx.x;
  const float* p = x + row * DIMF;
  int t = threadIdx.x;
  float v0 = p[t], v1 = p[t + 256];
  red[t] = v0 + v1; __syncthreads();
  for (int s = 128; s > 0; s >>= 1) { if (t < s) red[t] += red[t + s]; __syncthreads(); }
  float mu = red[0] * (1.0f / DIMF); __syncthreads();
  float d0 = v0 - mu, d1 = v1 - mu;
  red[t] = d0 * d0 + d1 * d1; __syncthreads();
  for (int s = 128; s > 0; s >>= 1) { if (t < s) red[t] += red[t + s]; __syncthreads(); }
  float rstd = rsqrtf(red[0] * (1.0f / DIMF) + 1e-5f);
  out[row * DIMF + t]       = (_Float16)(d0 * rstd * g[t]       + b[t]);
  out[row * DIMF + t + 256] = (_Float16)(d1 * rstd * g[t + 256] + b[t + 256]);
}

__global__ __launch_bounds__(256) void f32_to_f16_k(const float* in, _Float16* out, long n) {
  long i = (long)blockIdx.x * 256 + threadIdx.x;
  if (i < n) out[i] = (_Float16)in[i];
}

// out[c*R + r] = (f16) in[r*C + c]   (pre-transpose weights to NxK)
__global__ __launch_bounds__(256) void transpose_f32f16_k(const float* in, _Float16* out,
                                                          int R, int C) {
  long i = (long)blockIdx.x * 256 + threadIdx.x;
  if (i >= (long)R * C) return;
  int r = i / C, c = i % C;
  out[(long)c * R + r] = (_Float16)in[i];
}

// mean over groups of 16 tokens: [bh,4096,64] -> [bh,256,64]
__global__ __launch_bounds__(256) void landmark_pool_k(const _Float16* in, _Float16* out) {
  long gid = (long)blockIdx.x * 256 + threadIdx.x;
  if (gid >= (long)BHD * LMK * DH) return;
  int z = gid / (LMK * DH);
  int rem = gid % (LMK * DH);
  int mm = rem / DH, d = rem % DH;
  const _Float16* p = in + (long)z * NTOK * DH + (long)mm * 16 * DH + d;
  float s = 0.f;
#pragma unroll
  for (int i = 0; i < 16; ++i) s += (float)p[i * DH];
  out[gid] = (_Float16)(s * (1.0f / 16.0f));
}

// row softmax over f16 logits; f16 in place, or f32 to out_f32
__global__ __launch_bounds__(256) void softmax_rows_k(_Float16* io, float* out_f32, int rowlen) {
  __shared__ float red[256];
  long row = blockIdx.x;
  _Float16* p = io + row * (long)rowlen;
  int t = threadIdx.x, per = rowlen >> 8;
  float vals[16];
  float mx = -1e30f;
  for (int i = 0; i < per; ++i) { float v = (float)p[t + (i << 8)]; vals[i] = v; mx = fmaxf(mx, v); }
  red[t] = mx; __syncthreads();
  for (int s = 128; s > 0; s >>= 1) { if (t < s) red[t] = fmaxf(red[t], red[t + s]); __syncthreads(); }
  mx = red[0]; __syncthreads();
  float sum = 0.f;
  for (int i = 0; i < per; ++i) { vals[i] = __expf(vals[i] - mx); sum += vals[i]; }
  red[t] = sum; __syncthreads();
  for (int s = 128; s > 0; s >>= 1) { if (t < s) red[t] += red[t + s]; __syncthreads(); }
  float inv = 1.0f / red[0];
  if (out_f32) {
    float* q = out_f32 + row * (long)rowlen;
    for (int i = 0; i < per; ++i) q[t + (i << 8)] = vals[i] * inv;
  } else {
    for (int i = 0; i < per; ++i) p[t + (i << 8)] = (_Float16)(vals[i] * inv);
  }
}

// scale = 1/(max abs row-sum * max abs col-sum) per 256x256 matrix
__global__ __launch_bounds__(256) void pinv_scale_k(const float* a, float* scale) {
  __shared__ float r1[256], r2[256];
  const float* A = a + (long)blockIdx.x * 65536;
  int t = threadIdx.x;
  float rs = 0.f, cs = 0.f;
  for (int j = 0; j < 256; ++j) { rs += fabsf(A[t * 256 + j]); cs += fabsf(A[j * 256 + t]); }
  r1[t] = rs; r2[t] = cs; __syncthreads();
  for (int s = 128; s > 0; s >>= 1) {
    if (t < s) { r1[t] = fmaxf(r1[t], r1[t + s]); r2[t] = fmaxf(r2[t], r2[t + s]); }
    __syncthreads();
  }
  if (t == 0) scale[blockIdx.x] = 1.0f / (r1[0] * r2[0]);
}

// z = a^T * s  and  zT = a * s  (zT is z stored NxK for async-staged GEMM B)
__global__ __launch_bounds__(256) void pinv_init_z_k(const float* a, const float* scale,
                                                     float* z, float* zT) {
  long gid = (long)blockIdx.x * 256 + threadIdx.x;
  if (gid >= (long)BHD * 65536) return;
  int zb = gid >> 16;
  int r = (gid >> 8) & 255, c = gid & 255;
  float s = scale[zb];
  z[gid]  = a[((long)zb << 16) + (long)c * 256 + r] * s;
  zT[gid] = a[gid] * s;
}

// depthwise token conv (k=33, same pad) on v, add attention out, write
// concat f16 in [b, n, h*dh] layout for the final projection GEMM
__global__ __launch_bounds__(256) void conv_add_transpose_k(
    const float* __restrict__ oatt, const _Float16* __restrict__ v,
    const float* __restrict__ rw, _Float16* __restrict__ concat)
{
  long gid = (long)blockIdx.x * 256 + threadIdx.x;
  if (gid >= (long)BHD * NTOK * DH) return;
  int z = gid / (NTOK * DH);
  int rem = gid % (NTOK * DH);
  int n = rem / DH, d = rem % DH;
  int b = z / HEADS, h = z % HEADS;
  const _Float16* vb = v + (long)z * NTOK * DH;
  float res = 0.f;
#pragma unroll
  for (int kk = 0; kk < KERN; ++kk) {
    int tn = n + kk - (KERN / 2);
    if (tn >= 0 && tn < NTOK) res += rw[h * KERN + kk] * (float)vb[(long)tn * DH + d];
  }
  concat[((long)(b * NTOK + n)) * DIMF + h * DH + d] = (_Float16)(oatt[gid] + res);
}

// ===========================================================================
extern "C" void kernel_launch(void* const* d_in, const int* in_sizes, int n_in,
                              void* d_out, int out_size, void* d_ws, size_t ws_size,
                              hipStream_t stream) {
  const float* x     = (const float*)d_in[0];
  const float* w_qkv = (const float*)d_in[1];
  const float* w_out = (const float*)d_in[2];
  const float* b_out = (const float*)d_in[3];
  const float* res_w = (const float*)d_in[4];
  const float* ln_g  = (const float*)d_in[5];
  const float* ln_b  = (const float*)d_in[6];

  char* w = (char*)d_ws;
  size_t off = 0;
  auto alloc = [&](size_t bytes) { size_t r = off; off += (bytes + 255) & ~(size_t)255; return r; };
  const long TOKS = (long)BATCH * NTOK;
  _Float16* XN    = (_Float16*)(w + alloc(TOKS * DIMF * 2));
  _Float16* WQKVT = (_Float16*)(w + alloc((size_t)DIMF * 1536 * 2));
  _Float16* WOUTT = (_Float16*)(w + alloc((size_t)DIMF * DIMF * 2));
  _Float16* Q     = (_Float16*)(w + alloc((size_t)BHD * NTOK * DH * 2));
  _Float16* Kh    = (_Float16*)(w + alloc((size_t)BHD * NTOK * DH * 2));
  _Float16* V     = (_Float16*)(w + alloc((size_t)BHD * NTOK * DH * 2));
  _Float16* VT    = (_Float16*)(w + alloc((size_t)BHD * NTOK * DH * 2));
  _Float16* QL    = (_Float16*)(w + alloc((size_t)BHD * LMK * DH * 2));
  _Float16* KL    = (_Float16*)(w + alloc((size_t)BHD * LMK * DH * 2));
  _Float16* ATT1  = (_Float16*)(w + alloc((size_t)BHD * NTOK * LMK * 2));
  _Float16* LOG2  = (_Float16*)(w + alloc((size_t)BHD * LMK * LMK * 2));
  _Float16* ATT3  = (_Float16*)(w + alloc((size_t)BHD * LMK * NTOK * 2));
  float*    A2f   = (float*)   (w + alloc((size_t)BHD * LMK * LMK * 4));
  float*    ZA    = (float*)   (w + alloc((size_t)BHD * LMK * LMK * 4));
  float*    ZB    = (float*)   (w + alloc((size_t)BHD * LMK * LMK * 4));
  float*    ZAT   = (float*)   (w + alloc((size_t)BHD * LMK * LMK * 4));
  float*    ZBT   = (float*)   (w + alloc((size_t)BHD * LMK * LMK * 4));
  float*    AZ    = (float*)   (w + alloc((size_t)BHD * LMK * LMK * 4));
  float*    AZT   = (float*)   (w + alloc((size_t)BHD * LMK * LMK * 4));
  float*    TTT   = (float*)   (w + alloc((size_t)BHD * LMK * LMK * 4));
  float*    UUT   = (float*)   (w + alloc((size_t)BHD * LMK * LMK * 4));
  float*    SCL   = (float*)   (w + alloc(BHD * 4));
  _Float16* ZHT   = (_Float16*)(w + alloc((size_t)BHD * LMK * LMK * 2));
  _Float16* KVT   = (_Float16*)(w + alloc((size_t)BHD * LMK * DH * 2));
  _Float16* W1    = (_Float16*)(w + alloc((size_t)BHD * NTOK * LMK * 2));
  float*    OATT  = (float*)   (w + alloc((size_t)BHD * NTOK * DH * 4));
  _Float16* CONC  = (_Float16*)(w + alloc(TOKS * DIMF * 2));
  (void)ws_size; (void)in_sizes; (void)n_in; (void)out_size;

  // 1) LayerNorm -> f16; weights converted AND transposed to NxK
  layernorm_k<<<TOKS, 256, 0, stream>>>(x, ln_g, ln_b, XN);
  transpose_f32f16_k<<<3072, 256, 0, stream>>>(w_qkv, WQKVT, 512, 1536);
  transpose_f32f16_k<<<1024, 256, 0, stream>>>(w_out, WOUTT, 512, 512);

  // 2) QKV projection, fused head-split + q-scale + V^T emit
  gemm_f16_k<3><<<dim3(24, 128, 1), 256, 0, stream>>>(
      XN, WQKVT, 16384, 1536, 512, 0, 0, 0,
      nullptr, nullptr, nullptr, Q, Kh, V, VT, 0.125f);

  // 3) landmark pooling
  landmark_pool_k<<<2048, 256, 0, stream>>>(Q,  QL);
  landmark_pool_k<<<2048, 256, 0, stream>>>(Kh, KL);

  // 4) attention logits + softmax
  gemm_f16_k<0><<<dim3(4, 32, 32), 256, 0, stream>>>(      // S1 = q @ k_l^T
      Q, KL, 4096, 256, 64, 262144, 16384, 1048576,
      ATT1, nullptr, nullptr, nullptr, nullptr, nullptr, nullptr, 0.f);
  softmax_rows_k<<<131072, 256, 0, stream>>>(ATT1, nullptr, 256);
  gemm_f16_k<0><<<dim3(4, 2, 32), 256, 0, stream>>>(       // S2 = q_l @ k_l^T
      QL, KL, 256, 256, 64, 16384, 16384, 65536,
      LOG2, nullptr, nullptr, nullptr, nullptr, nullptr, nullptr, 0.f);
  softmax_rows_k<<<8192, 256, 0, stream>>>(LOG2, A2f, 256);
  gemm_f16_k<0><<<dim3(64, 2, 32), 256, 0, stream>>>(      // S3 = q_l @ k^T
      QL, Kh, 256, 4096, 64, 16384, 262144, 1048576,
      ATT3, nullptr, nullptr, nullptr, nullptr, nullptr, nullptr, 0.f);
  softmax_rows_k<<<8192, 256, 0, stream>>>(ATT3, nullptr, 4096);

  // 5) Moore-Penrose pinv (6 Newton-Schulz iterations, f32 WMMA, all GEMM
  //    B-operands pre-transposed so both tiles stage via async b128)
  pinv_scale_k<<<BHD, 256, 0, stream>>>(A2f, SCL);
  pinv_init_z_k<<<8192, 256, 0, stream>>>(A2f, SCL, ZA, ZAT);
  float* zc = ZA;  float* zcT = ZAT;
  float* zn = ZB;  float* znT = ZBT;
  dim3 g32(2, 4, BHD);
  for (int it = 0; it < 6; ++it) {
    gemm_f32_k<<<g32, 256, 0, stream>>>(A2f, zcT, nullptr, AZ, AZT, 1.0f, 0.0f);   // az
    gemm_f32_k<<<g32, 256, 0, stream>>>(AZ, AZT, AZ, nullptr, TTT, -1.0f, 7.0f);   // t = 7az - az^2
    gemm_f32_k<<<g32, 256, 0, stream>>>(AZ, TTT, AZ, nullptr, UUT, -1.0f, 15.0f);  // u = 15az - az@t
    gemm_f32_k<<<g32, 256, 0, stream>>>(zc, UUT, zc, zn, znT, -0.25f, 3.25f);      // z' = 3.25z - .25 z@u
    float* tp;
    tp = zc; zc = zn; zn = tp;
    tp = zcT; zcT = znT; znT = tp;
  }
  f32_to_f16_k<<<8192, 256, 0, stream>>>(zcT, ZHT, (long)BHD * 65536);  // Z^T in f16

  // 6) recombination: out = (attn1 @ pinv) @ (attn3 @ v)
  gemm_f16_k<4><<<dim3(1, 2, 32), 256, 0, stream>>>(       // KV^T = (attn3 @ v)^T
      ATT3, VT, 256, 64, 4096, 1048576, 262144, 16384,
      KVT, nullptr, nullptr, nullptr, nullptr, nullptr, nullptr, 0.f);
  gemm_f16_k<0><<<dim3(4, 32, 32), 256, 0, stream>>>(      // w1 = attn1 @ z
      ATT1, ZHT, 4096, 256, 256, 1048576, 65536, 1048576,
      W1, nullptr, nullptr, nullptr, nullptr, nullptr, nullptr, 0.f);
  gemm_f16_k<1><<<dim3(1, 32, 32), 256, 0, stream>>>(      // oatt = w1 @ kv (f32)
      W1, KVT, 4096, 64, 256, 1048576, 16384, 262144,
      OATT, nullptr, nullptr, nullptr, nullptr, nullptr, nullptr, 0.f);

  // 7) depthwise conv residual + transpose into [b,n,h*dh] f16
  conv_add_transpose_k<<<32768, 256, 0, stream>>>(OATT, V, res_w, CONC);

  // 8) final projection + bias + input residual -> d_out (f32)
  gemm_f16_k<2><<<dim3(8, 128, 1), 256, 0, stream>>>(
      CONC, WOUTT, 16384, 512, 512, 0, 0, 0,
      (float*)d_out, b_out, x, nullptr, nullptr, nullptr, nullptr, 0.f);
}
